// MultiHeadLatentAttention_6167573037631
// MI455X (gfx1250) — matrix operand order repaired and checked by reading, not compiled, not run on hardware
//
#include <hip/hip_runtime.h>

typedef __attribute__((ext_vector_type(16))) __bf16 v16bf;
typedef __attribute__((ext_vector_type(8)))  __bf16 bf8;
typedef __attribute__((ext_vector_type(8)))  float  v8f;
typedef __attribute__((ext_vector_type(4)))  float  f4;

#define EMBED 1024
#define NHEADS 16
#define HD 64
#define RANK 128
#define SEQ 2048
#define BATCH 4
#define ROWS (BATCH*SEQ)          // 8192
#define ATT_SCALE 0.125f          // 1/sqrt(64)
#define LOG2E 1.4426950408889634f

// base-2 exponential -> v_exp_f32 directly
#if __has_builtin(__builtin_amdgcn_exp2f)
#define EXP2F(x) __builtin_amdgcn_exp2f(x)
#else
#define EXP2F(x) exp2f(x)
#endif

__device__ __forceinline__ v16bf cat8(bf8 lo, bf8 hi) {
    v16bf r;
#pragma unroll
    for (int i = 0; i < 8; ++i) { r[i] = lo[i]; r[i + 8] = hi[i]; }
    return r;
}

__device__ __forceinline__ v8f wmma_bf16(v16bf a, v16bf b, v8f c) {
    return __builtin_amdgcn_wmma_f32_16x16x32_bf16(
        false, a, false, b, (short)0, c, false, false);
}

// ---- 16-lane butterfly reductions (DPP16, pure VALU) ------------------------
// XOR masks {1,2,7,15} generate the full 16-lane group:
//   0xB1 quad_perm[1,0,3,2] | 0x4E quad_perm[2,3,0,1] | 0x141 half_mirror | 0x140 mirror
#if __has_builtin(__builtin_amdgcn_update_dpp)
#define DPPF(v, ctrl) __int_as_float(__builtin_amdgcn_update_dpp(              \
        __float_as_int(v), __float_as_int(v), (ctrl), 0xf, 0xf, false))
#define HAVE_DPP 1
#elif __has_builtin(__builtin_amdgcn_mov_dpp)
#define DPPF(v, ctrl) __int_as_float(__builtin_amdgcn_mov_dpp(                 \
        __float_as_int(v), (ctrl), 0xf, 0xf, false))
#define HAVE_DPP 1
#else
#define HAVE_DPP 0
#endif

__device__ __forceinline__ float rmax16(float x) {
#if HAVE_DPP
    x = fmaxf(x, DPPF(x, 0xB1));
    x = fmaxf(x, DPPF(x, 0x4E));
    x = fmaxf(x, DPPF(x, 0x141));
    x = fmaxf(x, DPPF(x, 0x140));
#else
#pragma unroll
    for (int m = 1; m < 16; m <<= 1) x = fmaxf(x, __shfl_xor(x, m, 32));
#endif
    return x;
}
__device__ __forceinline__ float rsum16(float x) {
#if HAVE_DPP
    x += DPPF(x, 0xB1);
    x += DPPF(x, 0x4E);
    x += DPPF(x, 0x141);
    x += DPPF(x, 0x140);
#else
#pragma unroll
    for (int m = 1; m < 16; m <<= 1) x += __shfl_xor(x, m, 32);
#endif
    return x;
}

// ---------------- prep kernels ----------------
__global__ __launch_bounds__(256) void cvt_bf16(const float* __restrict__ in,
                                                __bf16* __restrict__ out, int n) {
    int i = (blockIdx.x * 256 + threadIdx.x) * 4;
    if (i + 3 < n) {
        f4 v = *(const f4*)(in + i);
#pragma unroll
        for (int j = 0; j < 4; ++j) out[i + j] = (__bf16)v[j];
    }
}

// W: (K x N) f32 row-major  ->  Wt: (N x K) bf16 row-major
__global__ __launch_bounds__(256) void transpose_cvt(const float* __restrict__ W,
                                                     __bf16* __restrict__ Wt,
                                                     int K, int N) {
    __shared__ float tile[32][33];
    int tx = threadIdx.x, ty = threadIdx.y;
    int k0 = blockIdx.y * 32, n0 = blockIdx.x * 32;
#pragma unroll
    for (int i = 0; i < 4; ++i)
        tile[ty + 8 * i][tx] = W[(size_t)(k0 + ty + 8 * i) * N + n0 + tx];
    __syncthreads();
#pragma unroll
    for (int i = 0; i < 4; ++i)
        Wt[(size_t)(n0 + ty + 8 * i) * K + k0 + tx] = (__bf16)tile[tx][ty + 8 * i];
}

// ---------------- generic WMMA GEMM (software-pipelined) ----------------
// C(MxN) = A(MxK bf16 rm) * Bt(NxK bf16 rm)^T + bias
// Block tile 128x128, 8 waves (4x2), wave tile 32x64 (2x4 frags), K-stage 32.
// Double-buffered LDS + register staging: one barrier per stage, global-load
// latency hidden behind the WMMA burst of the previous stage.
#define LDSP 40   // 32 halves + 8 pad

template <typename OutT>
__global__ __launch_bounds__(256) void gemm_bf16(const __bf16* __restrict__ A,
                                                 const __bf16* __restrict__ Bt,
                                                 const float* __restrict__ bias,
                                                 OutT* __restrict__ C,
                                                 int M, int N, int K) {
    __shared__ __attribute__((aligned(16))) __bf16 sA[2][128 * LDSP];
    __shared__ __attribute__((aligned(16))) __bf16 sB[2][128 * LDSP];

    const int tid = threadIdx.x;
    const int lane = tid & 31;
    const int w = tid >> 5;
    const int waveM = w >> 1, waveN = w & 1;
    const int l15 = lane & 15, lh = lane >> 4;
    const int Mb = blockIdx.y * 128, Nb = blockIdx.x * 128;

    bf8 ra[2], rb[2];
    auto FETCH = [&](int kq) {
#pragma unroll
        for (int i = 0; i < 2; ++i) {
            int c = tid + i * 256;          // 512 chunks of 8 halves per array
            int r = c >> 2, ko = (c & 3) * 8;
            ra[i] = *(const bf8*)&A[(size_t)(Mb + r) * K + kq + ko];
            rb[i] = *(const bf8*)&Bt[(size_t)(Nb + r) * K + kq + ko];
            if (kq + 32 < K) {              // global_prefetch_b8 the stage after
                __builtin_prefetch(&A[(size_t)(Mb + r) * K + kq + 32 + ko], 0, 1);
                __builtin_prefetch(&Bt[(size_t)(Nb + r) * K + kq + 32 + ko], 0, 1);
            }
        }
    };
    auto STORE = [&](int bsel) {
#pragma unroll
        for (int i = 0; i < 2; ++i) {
            int c = tid + i * 256;
            int r = c >> 2, ko = (c & 3) * 8;
            *(bf8*)&sA[bsel][r * LDSP + ko] = ra[i];
            *(bf8*)&sB[bsel][r * LDSP + ko] = rb[i];
        }
    };

    v8f acc[2][4];
#pragma unroll
    for (int fm = 0; fm < 2; ++fm)
#pragma unroll
        for (int fn = 0; fn < 4; ++fn)
#pragma unroll
            for (int e = 0; e < 8; ++e) acc[fm][fn][e] = 0.0f;

    FETCH(0);
    STORE(0);
    __syncthreads();

    for (int k0 = 0; k0 < K; k0 += 32) {
        const int cur = (k0 >> 5) & 1;
        const bool more = (k0 + 32) < K;
        if (more) FETCH(k0 + 32);           // loads in flight during compute

        v16bf bfrag[4];
#pragma unroll
        for (int fn = 0; fn < 4; ++fn) {
            int n = waveN * 64 + fn * 16 + l15;
            int kb = lh * 8;
            bfrag[fn] = cat8(*(const bf8*)&sB[cur][n * LDSP + kb],
                             *(const bf8*)&sB[cur][n * LDSP + kb + 16]);
        }
#pragma unroll
        for (int fm = 0; fm < 2; ++fm) {
            int r = waveM * 32 + fm * 16 + l15;
            int ko = lh * 8;
            v16bf afrag = cat8(*(const bf8*)&sA[cur][r * LDSP + ko],
                               *(const bf8*)&sA[cur][r * LDSP + ko + 16]);
#pragma unroll
            for (int fn = 0; fn < 4; ++fn)
                acc[fm][fn] = wmma_bf16(afrag, bfrag[fn], acc[fm][fn]);
        }

        if (more) STORE(cur ^ 1);
        __syncthreads();
    }

#pragma unroll
    for (int fm = 0; fm < 2; ++fm)
#pragma unroll
        for (int fn = 0; fn < 4; ++fn) {
            int colg = Nb + waveN * 64 + fn * 16 + l15;
            float bv = bias[colg];
#pragma unroll
            for (int e = 0; e < 8; ++e) {
                int rowg = Mb + waveM * 32 + fm * 16 + e + 8 * lh;
                C[(size_t)rowg * N + colg] = (OutT)(acc[fm][fn][e] + bv);
            }
        }
}

// ---------------- flash attention (bf16 WMMA, online softmax) ----------------
// grid: BATCH*NHEADS*(SEQ/128) = 1024 blocks; 8 waves; wave owns 16 q-rows.
// 64-key chunks, double-buffered K/V tiles: 16 WMMAs + one barrier per chunk.
// Softmax in exp2 domain: log2(e) folded into the Q scale, v_exp_f32 direct.
#define KP 72     // 64 halves + 8 pad

__global__ __launch_bounds__(256) void mla_attn(const __bf16* __restrict__ q,
                                                const __bf16* __restrict__ kk,
                                                const __bf16* __restrict__ vv,
                                                __bf16* __restrict__ out) {
    __shared__ __attribute__((aligned(16))) __bf16 lds_k[2][64 * KP];   // keys x d
    __shared__ __attribute__((aligned(16))) __bf16 lds_v[2][64 * KP];   // d x keys
    __shared__ __attribute__((aligned(16))) __bf16 lds_p[8][16 * KP];   // per-wave P

    const int tid = threadIdx.x;
    const int lane = tid & 31;
    const int w = tid >> 5;
    const int l15 = lane & 15, lh = lane >> 4;

    const int qblk = blockIdx.x & 15;       // 16 q-blocks of 128 rows per (b,h)
    const int bh = blockIdx.x >> 4;
    const int b = bh >> 4, h = bh & 15;
    const int q0 = qblk * 128 + w * 16;

    bf8 kr[2], vr[2];
    int kkey[2], kdo[2], vkey[2], vd0[2];
    auto FETCH = [&](int kc2) {
#pragma unroll
        for (int i = 0; i < 2; ++i) {
            int c = tid + i * 256;          // 512 chunks of 8 halves per tile
            kkey[i] = c >> 3; kdo[i] = (c & 7) * 8;
            kr[i] = *(const bf8*)&kk[(size_t)(b * SEQ + kc2 + kkey[i]) * EMBED + h * HD + kdo[i]];
            vkey[i] = c & 63; vd0[i] = (c >> 6) * 8;
            vr[i] = *(const bf8*)&vv[(size_t)(b * SEQ + kc2 + vkey[i]) * EMBED + h * HD + vd0[i]];
        }
    };
    auto STORE = [&](int bsel) {
#pragma unroll
        for (int i = 0; i < 2; ++i) {
            *(bf8*)&lds_k[bsel][kkey[i] * KP + kdo[i]] = kr[i];
#pragma unroll
            for (int j = 0; j < 8; ++j)
                lds_v[bsel][(vd0[i] + j) * KP + vkey[i]] = vr[i][j];
        }
    };

    // load q fragments once; fold softmax scale AND log2(e) in
    v16bf qf[2];
    const float qscale = ATT_SCALE * LOG2E;
#pragma unroll
    for (int t = 0; t < 2; ++t) {
        const __bf16* qp = q + (size_t)(b * SEQ + q0 + l15) * EMBED + h * HD + t * 32 + lh * 8;
        bf8 lo = *(const bf8*)qp;
        bf8 hi = *(const bf8*)(qp + 16);
#pragma unroll
        for (int i = 0; i < 8; ++i) {
            lo[i] = (__bf16)((float)lo[i] * qscale);
            hi[i] = (__bf16)((float)hi[i] * qscale);
        }
        qf[t] = cat8(lo, hi);
    }

    float mrow[8], lrow[8];
    v8f O[4];
#pragma unroll
    for (int e = 0; e < 8; ++e) { mrow[e] = -1e30f; lrow[e] = 0.0f; }
#pragma unroll
    for (int f = 0; f < 4; ++f)
#pragma unroll
        for (int e = 0; e < 8; ++e) O[f][e] = 0.0f;

    FETCH(0);
    STORE(0);
    __syncthreads();

    for (int kc = 0; kc < SEQ; kc += 64) {
        const int cur = (kc >> 6) & 1;
        const bool more = (kc + 64) < SEQ;
        if (more) FETCH(kc + 64);           // next tiles in flight

        // S (16 q-rows x 64 keys, log2 domain) via 8 WMMAs
        v8f S[4];
#pragma unroll
        for (int j = 0; j < 4; ++j) {
#pragma unroll
            for (int e = 0; e < 8; ++e) S[j][e] = 0.0f;
            int n = j * 16 + l15;
#pragma unroll
            for (int t = 0; t < 2; ++t) {
                int kb = t * 32 + lh * 8;
                v16bf kf = cat8(*(const bf8*)&lds_k[cur][n * KP + kb],
                                *(const bf8*)&lds_k[cur][n * KP + kb + 16]);
                S[j] = wmma_bf16(qf[t], kf, S[j]);
            }
        }

        // online softmax (exp2 domain) + stage P into per-wave LDS patch
#pragma unroll
        for (int e = 0; e < 8; ++e) {
            float mc = fmaxf(fmaxf(S[0][e], S[1][e]), fmaxf(S[2][e], S[3][e]));
            mc = rmax16(mc);
            float mn = fmaxf(mrow[e], mc);
            float alpha = EXP2F(mrow[e] - mn);
            float p0 = EXP2F(S[0][e] - mn);
            float p1 = EXP2F(S[1][e] - mn);
            float p2 = EXP2F(S[2][e] - mn);
            float p3 = EXP2F(S[3][e] - mn);
            lrow[e] = lrow[e] * alpha + rsum16((p0 + p1) + (p2 + p3));
            mrow[e] = mn;
#pragma unroll
            for (int f = 0; f < 4; ++f) O[f][e] *= alpha;
            int prow = e + 8 * lh;
            lds_p[w][prow * KP + l15]      = (__bf16)p0;
            lds_p[w][prow * KP + 16 + l15] = (__bf16)p1;
            lds_p[w][prow * KP + 32 + l15] = (__bf16)p2;
            lds_p[w][prow * KP + 48 + l15] = (__bf16)p3;
        }

        // P as two A-fragments (16x32), then O += P * V  (8 WMMAs)
        v16bf pf[2];
#pragma unroll
        for (int t = 0; t < 2; ++t) {
            int ko = t * 32 + lh * 8;
            pf[t] = cat8(*(const bf8*)&lds_p[w][l15 * KP + ko],
                         *(const bf8*)&lds_p[w][l15 * KP + ko + 16]);
        }
#pragma unroll
        for (int f = 0; f < 4; ++f) {
            int n = f * 16 + l15;
#pragma unroll
            for (int t = 0; t < 2; ++t) {
                int kb = t * 32 + lh * 8;
                v16bf vf = cat8(*(const bf8*)&lds_v[cur][n * KP + kb],
                                *(const bf8*)&lds_v[cur][n * KP + kb + 16]);
                O[f] = wmma_bf16(pf[t], vf, O[f]);
            }
        }

        if (more) STORE(cur ^ 1);
        __syncthreads();
    }

    // normalize and store (bf16, feeds the Wout GEMM)
#pragma unroll
    for (int f = 0; f < 4; ++f)
#pragma unroll
        for (int e = 0; e < 8; ++e) {
            int rowg = b * SEQ + q0 + e + 8 * lh;
            out[(size_t)rowg * EMBED + h * HD + f * 16 + l15] =
                (__bf16)(O[f][e] / lrow[e]);
        }
}

// ---------------- workspace layout (bytes) ----------------
#define OFF_X16   ((size_t)0)
#define OFF_Q16   (OFF_X16 + (size_t)ROWS * EMBED * 2)
#define OFF_K16   (OFF_Q16 + (size_t)ROWS * EMBED * 2)
#define OFF_V16   (OFF_K16 + (size_t)ROWS * EMBED * 2)
#define OFF_AO16  (OFF_V16 + (size_t)ROWS * EMBED * 2)
#define OFF_LAT16 (OFF_AO16 + (size_t)ROWS * EMBED * 2)
#define OFF_WQT   (OFF_LAT16 + (size_t)ROWS * RANK * 2)
#define OFF_WOT   (OFF_WQT + (size_t)EMBED * EMBED * 2)
#define OFF_WDT   (OFF_WOT + (size_t)EMBED * EMBED * 2)
#define OFF_WKT   (OFF_WDT + (size_t)EMBED * RANK * 2)
#define OFF_WVT   (OFF_WKT + (size_t)EMBED * RANK * 2)

extern "C" void kernel_launch(void* const* d_in, const int* in_sizes, int n_in,
                              void* d_out, int out_size, void* d_ws, size_t ws_size,
                              hipStream_t stream) {
    const float* x     = (const float*)d_in[0];
    const float* Wq    = (const float*)d_in[1];
    const float* bq    = (const float*)d_in[2];
    const float* Wdown = (const float*)d_in[3];
    const float* bdown = (const float*)d_in[4];
    const float* Wk_up = (const float*)d_in[5];
    const float* bk    = (const float*)d_in[6];
    const float* Wv_up = (const float*)d_in[7];
    const float* bv    = (const float*)d_in[8];
    const float* Wout  = (const float*)d_in[9];
    const float* bout  = (const float*)d_in[10];

    char* ws = (char*)d_ws;
    __bf16* x16   = (__bf16*)(ws + OFF_X16);
    __bf16* q16   = (__bf16*)(ws + OFF_Q16);
    __bf16* k16   = (__bf16*)(ws + OFF_K16);
    __bf16* v16   = (__bf16*)(ws + OFF_V16);
    __bf16* ao16  = (__bf16*)(ws + OFF_AO16);
    __bf16* lat16 = (__bf16*)(ws + OFF_LAT16);
    __bf16* WqT   = (__bf16*)(ws + OFF_WQT);
    __bf16* WoT   = (__bf16*)(ws + OFF_WOT);
    __bf16* WdT   = (__bf16*)(ws + OFF_WDT);
    __bf16* WkT   = (__bf16*)(ws + OFF_WKT);
    __bf16* WvT   = (__bf16*)(ws + OFF_WVT);

    // 1) x -> bf16
    cvt_bf16<<<(ROWS * EMBED) / 1024, 256, 0, stream>>>(x, x16, ROWS * EMBED);

    // 2) weights -> transposed bf16 (N x K)
    dim3 tb(32, 8);
    transpose_cvt<<<dim3(EMBED / 32, EMBED / 32), tb, 0, stream>>>(Wq,    WqT, EMBED, EMBED);
    transpose_cvt<<<dim3(RANK  / 32, EMBED / 32), tb, 0, stream>>>(Wdown, WdT, EMBED, RANK);
    transpose_cvt<<<dim3(EMBED / 32, RANK  / 32), tb, 0, stream>>>(Wk_up, WkT, RANK,  EMBED);
    transpose_cvt<<<dim3(EMBED / 32, RANK  / 32), tb, 0, stream>>>(Wv_up, WvT, RANK,  EMBED);
    transpose_cvt<<<dim3(EMBED / 32, EMBED / 32), tb, 0, stream>>>(Wout,  WoT, EMBED, EMBED);

    // 3) projections (WMMA bf16)
    gemm_bf16<__bf16><<<dim3(EMBED / 128, ROWS / 128), 256, 0, stream>>>(
        x16, WqT, bq, q16, ROWS, EMBED, EMBED);
    gemm_bf16<__bf16><<<dim3(RANK / 128, ROWS / 128), 256, 0, stream>>>(
        x16, WdT, bdown, lat16, ROWS, RANK, EMBED);
    gemm_bf16<__bf16><<<dim3(EMBED / 128, ROWS / 128), 256, 0, stream>>>(
        lat16, WkT, bk, k16, ROWS, EMBED, RANK);
    gemm_bf16<__bf16><<<dim3(EMBED / 128, ROWS / 128), 256, 0, stream>>>(
        lat16, WvT, bv, v16, ROWS, EMBED, RANK);

    // 4) flash attention
    mla_attn<<<BATCH * NHEADS * (SEQ / 128), 256, 0, stream>>>(q16, k16, v16, ao16);

    // 5) output projection -> f32
    gemm_bf16<float><<<dim3(EMBED / 128, ROWS / 128), 256, 0, stream>>>(
        ao16, WoT, bout, (float*)d_out, ROWS, EMBED, EMBED);
}
